// DynamicWeights__635655160217
// MI455X (gfx1250) — matrix operand order repaired and checked by reading, not compiled, not run on hardware
//
#include <hip/hip_runtime.h>
#include <hip/hip_bf16.h>

typedef __attribute__((ext_vector_type(16))) __bf16 bf16x16;
typedef __attribute__((ext_vector_type(8)))  float  v8f;
typedef __attribute__((ext_vector_type(4)))  unsigned int u32x4;

union ABFrag { u32x4 u[2]; bf16x16 v; };

// ---- bf16 helpers (RNE, branch-free) ----
__device__ __forceinline__ unsigned f2bf(float f) {
  union { float f; unsigned u; } x; x.f = f;
  unsigned r = x.u + 0x7FFFu + ((x.u >> 16) & 1u);
  return r >> 16;
}
__device__ __forceinline__ float bf2f(unsigned lo16) {
  union { unsigned u; float f; } x; x.u = lo16 << 16; return x.f;
}

// geometry
#define NB    32
#define CCH   256
#define HH    32
#define WW    32
#define HWP   1024         // 32*32
#define CIN2  512          // 2*C
#define OCH   18           // offset channels
#define KOFFC 400          // 25 taps * 16 c-groups  (K=12800 / 32)
#define KDEFC 72           // 9 taps * 8 c-groups    (K=2304  / 32)

// padded spatial: 36x36 (halo of 2, zero-filled)
#define PW    36
#define PPIX  1296                       // 36*36
#define NSTR  (PPIX * CIN2)              // elements per image = 663552

// ws layout (bytes)
#define INPT_BYTES  (42467328u)          // 32*1296*512*2  padded NHWC bf16
#define BOFF_BYTES  (819200u)            // 2*400*32*8*4
#define BD_BYTES    (73728u)             // 72*32*8*4

// ------------------------------------------------------------------
// Kernel 0: zero the padded input buffer (halo must be 0).
// ------------------------------------------------------------------
__global__ __launch_bounds__(256) void zero_inpT(u32x4* __restrict__ p) {
  size_t i = (size_t)blockIdx.x * 256 + threadIdx.x;   // 2,654,208 vec16 stores
  u32x4 z = {};
  p[i] = z;
}

// ------------------------------------------------------------------
// Kernel 1: NCHW f32 -> padded NHWC bf16 of concat(query, support)
// ------------------------------------------------------------------
__global__ __launch_bounds__(256) void pack_inpT(const float* __restrict__ support,
                                                 const float* __restrict__ query,
                                                 unsigned short* __restrict__ inpT) {
  int b = blockIdx.x;
  int y  = b & 31;
  int ct = (b >> 5) & 15;
  int n  = b >> 9;
  int c0 = ct * 32;
  __shared__ float tile[32][33];
  int t = threadIdx.x;
  int x  = t & 31;
  int cl = t >> 5;
#pragma unroll
  for (int p = 0; p < 4; ++p) {
    int c = c0 + cl + p * 8;
    float v;
    if (c < CCH) v = query  [(((size_t)n * CCH + c)         * HWP) + y * 32 + x];
    else         v = support[(((size_t)n * CCH + (c - CCH)) * HWP) + y * 32 + x];
    tile[cl + p * 8][x] = v;
  }
  __syncthreads();
  int cl2 = t & 31;
  int x2  = t >> 5;
#pragma unroll
  for (int p = 0; p < 4; ++p) {
    int xx = x2 + p * 8;
    inpT[((size_t)n * NSTR) + ((y + 2) * PW + (xx + 2)) * CIN2 + c0 + cl2] =
        (unsigned short)f2bf(tile[cl2][xx]);
  }
}

// ------------------------------------------------------------------
// Kernel 2: w_off (18,512,5,5) -> B-frag bf16 pairs.
// B frag: lane L -> col n=L&15 ; VGPR i -> K = (L>>4)*16 + 2i, +1
// ------------------------------------------------------------------
__global__ __launch_bounds__(256) void build_boff(const float* __restrict__ w_off,
                                                  unsigned int* __restrict__ Boff) {
  int idx = blockIdx.x * 256 + threadIdx.x;
  if (idx >= 2 * KOFFC * 32 * 8) return;
  int i  = idx & 7;
  int L  = (idx >> 3) & 31;
  int cc = (idx >> 8) % KOFFC;
  int ot = (idx >> 8) / KOFFC;
  int o  = ot * 16 + (L & 15);
  int K0 = ((L >> 4) * 16) + 2 * i;
  int tp = cc / 16, cg = cc % 16;
  int c  = cg * 32 + K0;
  unsigned val = 0;
  if (o < OCH) {
    float w0 = w_off[((size_t)(o * CIN2 + c    ) * 25) + tp];
    float w1 = w_off[((size_t)(o * CIN2 + c + 1) * 25) + tp];
    val = f2bf(w0) | (f2bf(w1) << 16);
  }
  Boff[idx] = val;
}

// ------------------------------------------------------------------
// Kernel 3: w_def (9,256,3,3) -> B frag ; kk = k*256 + c
// ------------------------------------------------------------------
__global__ __launch_bounds__(256) void build_bd(const float* __restrict__ w_def,
                                                unsigned int* __restrict__ Bd) {
  int idx = blockIdx.x * 256 + threadIdx.x;
  if (idx >= KDEFC * 32 * 8) return;
  int i  = idx & 7;
  int L  = (idx >> 3) & 31;
  int cc = idx >> 8;
  int o  = L & 15;
  int K0 = ((L >> 4) * 16) + 2 * i;
  int k  = cc >> 3;
  int c  = (cc & 7) * 32 + K0;
  unsigned val = 0;
  if (o < 9) {
    float w0 = w_def[(size_t)(o * CCH + c    ) * 9 + k];
    float w1 = w_def[(size_t)(o * CCH + c + 1) * 9 + k];
    val = f2bf(w0) | (f2bf(w1) << 16);
  }
  Bd[idx] = val;
}

// ------------------------------------------------------------------
// Kernel 4: offset 5x5 conv via bf16 WMMA implicit GEMM.
// 1 wave per 16-pixel tile; B fragments software-pipelined one chunk
// ahead so WMMAs of chunk i overlap loads of chunk i+1.
// ------------------------------------------------------------------
__global__ __launch_bounds__(256) void offset_conv(const unsigned short* __restrict__ inpT,
                                                   const unsigned int* __restrict__ Boff,
                                                   float* __restrict__ offout) {
  int wv   = blockIdx.x * 8 + (threadIdx.x >> 5);   // tile id 0..2047
  int L    = threadIdx.x & 31;
  int n    = wv >> 6;
  int pix0 = (wv & 63) << 4;
  int m    = L & 15;
  int kb   = (L >> 4) * 16;
  int pix  = pix0 + m;
  int y = pix >> 5, x = pix & 31;
  const unsigned short* base = inpT + (size_t)n * NSTR;
  const unsigned int*  bbase = Boff + (size_t)L * 8;
  v8f acc0 = {}; v8f acc1 = {};

  // prologue: B fragments for chunk 0
  ABFrag b0c, b1c;
  b0c.u[0] = *(const u32x4*)(bbase);
  b0c.u[1] = *(const u32x4*)(bbase + 4);
  b1c.u[0] = *(const u32x4*)(bbase + (size_t)KOFFC * 256);
  b1c.u[1] = *(const u32x4*)(bbase + (size_t)KOFFC * 256 + 4);

  for (int tp = 0; tp < 25; ++tp) {
    // padded coords: (y + dy + 2) = y + tp/5 ; same for x
    unsigned row = (unsigned)(((y + tp / 5) * PW + (x + tp % 5)) * CIN2);
#pragma unroll 4
    for (int cg = 0; cg < 16; ++cg) {
      unsigned c0 = row + (unsigned)(cg * 32 + kb);
      ABFrag a;
      a.u[0] = *(const u32x4*)(base + c0);
      a.u[1] = *(const u32x4*)(base + c0 + 8);

      int cc  = tp * 16 + cg;
      int ccn = (cc + 1 < KOFFC) ? (cc + 1) : (KOFFC - 1);   // uniform scalar guard
      ABFrag b0n, b1n;
      const unsigned int* bp0 = bbase + (size_t)ccn * 256;
      const unsigned int* bp1 = bbase + (size_t)(KOFFC + ccn) * 256;
      b0n.u[0] = *(const u32x4*)bp0; b0n.u[1] = *(const u32x4*)(bp0 + 4);
      b1n.u[0] = *(const u32x4*)bp1; b1n.u[1] = *(const u32x4*)(bp1 + 4);

      acc0 = __builtin_amdgcn_wmma_f32_16x16x32_bf16(false, a.v, false, b0c.v,
                                                     (short)0, acc0, false, false);
      acc1 = __builtin_amdgcn_wmma_f32_16x16x32_bf16(false, a.v, false, b1c.v,
                                                     (short)0, acc1, false, false);
      b0c = b0n; b1c = b1n;
    }
  }
  // D layout: vgpr r, lanes0-15: M=r, lanes16-31: M=8+r ; N = lane&15
  int half = L >> 4, oc = L & 15;
#pragma unroll
  for (int r = 0; r < 8; ++r) {
    int mm = r + 8 * half;
    offout[((size_t)(n * OCH + oc)) * HWP + pix0 + mm] = acc0[r];
    int o1 = 16 + oc;
    if (o1 < OCH)
      offout[((size_t)(n * OCH + o1)) * HWP + pix0 + mm] = acc1[r];
  }
}

// ------------------------------------------------------------------
// Kernel 5: deformable sampling + dynamic-weight GEMM + sigmoid.
// ------------------------------------------------------------------
__global__ __launch_bounds__(256) void deform_wmma(const unsigned short* __restrict__ inpT,
                                                   const unsigned int* __restrict__ Bd,
                                                   const float* __restrict__ offs,
                                                   float* __restrict__ out) {
  __shared__ int   s_addr[144][4];
  __shared__ float s_w[144][4];
  __shared__ float s_red[8][256];
  int tile = blockIdx.x;
  int n = tile >> 6;
  int pix0 = (tile & 63) << 4;
  int t = threadIdx.x;
  const unsigned short* base = inpT + (size_t)n * NSTR + CCH;   // support columns

  if (t < 144) {                      // (m,k) corner precompute
    int m = t / 9, k = t % 9;
    int pix = pix0 + m;
    int y = pix >> 5, x = pix & 31;
    float dy = offs[((size_t)(n * OCH + 2 * k    )) * HWP + pix];
    float dx = offs[((size_t)(n * OCH + 2 * k + 1)) * HWP + pix];
    float py = (float)y + (float)(k / 3 - 1) + dy;
    float px = (float)x + (float)(k % 3 - 1) + dx;
    float y0f = floorf(py), x0f = floorf(px);
    float wy1 = py - y0f, wy0 = 1.f - wy1;
    float wx1 = px - x0f, wx0 = 1.f - wx1;
    int y0 = (int)y0f, x0 = (int)x0f;
    int ys[2] = { y0, y0 + 1 }, xs[2] = { x0, x0 + 1 };
    float wy[2] = { wy0, wy1 }, wx[2] = { wx0, wx1 };
#pragma unroll
    for (int ci = 0; ci < 4; ++ci) {
      int yy = ys[ci >> 1], xx = xs[ci & 1];
      bool v = (yy >= 0) & (yy < HH) & (xx >= 0) & (xx < WW);
      int yc = min(max(yy, 0), 31), xc = min(max(xx, 0), 31);
      s_addr[t][ci] = ((yc + 2) * PW + (xc + 2)) * CIN2;   // padded coords
      s_w[t][ci] = v ? wy[ci >> 1] * wx[ci & 1] : 0.f;
    }
  }
  __syncthreads();

  int wvid = t >> 5, L = t & 31;
  int mL = L & 15, kb = (L >> 4) * 16;
  v8f acc = {};
  for (int j = 0; j < 9; ++j) {
    int cc = wvid * 9 + j;            // K-chunk 0..71
    int k  = cc >> 3;
    int c0 = (cc & 7) * 32;
    int e  = mL * 9 + k;
    int a0 = s_addr[e][0], a1 = s_addr[e][1], a2 = s_addr[e][2], a3 = s_addr[e][3];
    float w0 = s_w[e][0], w1 = s_w[e][1], w2 = s_w[e][2], w3 = s_w[e][3];
    ABFrag a;
#pragma unroll
    for (int seg = 0; seg < 2; ++seg) {
      int coff = c0 + kb + seg * 8;
      u32x4 v0 = *(const u32x4*)(base + a0 + coff);
      u32x4 v1 = *(const u32x4*)(base + a1 + coff);
      u32x4 v2 = *(const u32x4*)(base + a2 + coff);
      u32x4 v3 = *(const u32x4*)(base + a3 + coff);
      u32x4 pk;
#pragma unroll
      for (int d = 0; d < 4; ++d) {
        float lo = w0 * bf2f(v0[d] & 0xffffu) + w1 * bf2f(v1[d] & 0xffffu) +
                   w2 * bf2f(v2[d] & 0xffffu) + w3 * bf2f(v3[d] & 0xffffu);
        float hi = w0 * bf2f(v0[d] >> 16) + w1 * bf2f(v1[d] >> 16) +
                   w2 * bf2f(v2[d] >> 16) + w3 * bf2f(v3[d] >> 16);
        pk[d] = f2bf(lo) | (f2bf(hi) << 16);
      }
      a.u[seg] = pk;
    }
    ABFrag b;
    const unsigned int* bp = Bd + ((size_t)cc * 32 + L) * 8;
    b.u[0] = *(const u32x4*)bp; b.u[1] = *(const u32x4*)(bp + 4);
    acc = __builtin_amdgcn_wmma_f32_16x16x32_bf16(false, a.v, false, b.v,
                                                  (short)0, acc, false, false);
  }
#pragma unroll
  for (int r = 0; r < 8; ++r) s_red[wvid][r * 32 + L] = acc[r];
  __syncthreads();
  {
    int r = t >> 5, LL = t & 31;
    float s = 0.f;
#pragma unroll
    for (int ww = 0; ww < 8; ++ww) s += s_red[ww][r * 32 + LL];
    int mm = r + 8 * (LL >> 4);
    int o  = LL & 15;
    if (o < 9) {
      float val = 1.f / (1.f + __expf(-s));
      out[((size_t)(n * 9 + o)) * HWP + pix0 + mm] = val;
    }
  }
}

// ------------------------------------------------------------------
extern "C" void kernel_launch(void* const* d_in, const int* in_sizes, int n_in,
                              void* d_out, int out_size, void* d_ws, size_t ws_size,
                              hipStream_t stream) {
  (void)in_sizes; (void)n_in; (void)out_size; (void)ws_size;
  const float* support = (const float*)d_in[0];
  const float* query   = (const float*)d_in[1];
  const float* w_off   = (const float*)d_in[2];
  const float* w_def   = (const float*)d_in[3];
  float* out = (float*)d_out;

  char* ws = (char*)d_ws;
  unsigned short* inpT = (unsigned short*)ws;
  unsigned int*   Boff = (unsigned int*)(ws + INPT_BYTES);
  unsigned int*   Bd   = (unsigned int*)(ws + INPT_BYTES + BOFF_BYTES);
  float*          offs = (float*)(ws + INPT_BYTES + BOFF_BYTES + BD_BYTES);

  zero_inpT  <<<INPT_BYTES / 16 / 256, 256, 0, stream>>>((u32x4*)inpT);
  pack_inpT  <<<32 * 16 * 32,          256, 0, stream>>>(support, query, inpT);
  build_boff <<<800,                   256, 0, stream>>>(w_off, Boff);
  build_bd   <<<72,                    256, 0, stream>>>(w_def, Bd);
  offset_conv<<<256,                   256, 0, stream>>>(inpT, Boff, offs);
  deform_wmma<<<2048,                  256, 0, stream>>>(inpT, Bd, offs, out);
}
